// GraphConvolutionalLayer_7507602833631
// MI455X (gfx1250) — compile-verified, matrix operand order checked by reading, not run on hardware
//
#include <hip/hip_runtime.h>
#include <hip/hip_bf16.h>

// ---------------------------------------------------------------------------
// GCN layer for MI455X (gfx1250, wave32):
//   agg = segment_sum(edge_val * x[edge_src], edge_dst)   (atomic scatter, f32)
//   out = relu(agg @ W + b)                               (bf16 WMMA, f32 acc)
//
// GEMM: W pre-packed fragment-major bf16. Each block is pinned to one N-half
// (8 column tiles, 64 KB of packed W staged in LDS); each of its 8 waves owns
// one 16-row strip and accumulates 8 C-tiles (64 acc VGPRs -> no spills).
// A fragment is loaded/converted once per K-step and feeds 8 v_wmma ops.
// ---------------------------------------------------------------------------

#define D_FEAT 256
#define N_TILES_N (D_FEAT / 16)          // 16
#define TN_PER_WAVE 8                    // column tiles per wave (one N-half)
#define N_KSTEPS  (D_FEAT / 32)          // 8
#define W_PACK_ELEMS (N_TILES_N * N_KSTEPS * 32 * 16)     // 65536 bf16
#define W_HALF_ELEMS (W_PACK_ELEMS / 2)                   // 32768 bf16 = 64 KB

typedef __attribute__((ext_vector_type(16))) __bf16 v16bf;
typedef __attribute__((ext_vector_type(8)))  float  v8f;

__device__ __forceinline__ unsigned short f32_to_bf16_rne(float f) {
    unsigned int u = __float_as_uint(f);
    u += 0x7FFFu + ((u >> 16) & 1u);     // round-to-nearest-even
    return (unsigned short)(u >> 16);
}

// --- zero the accumulator workspace (float4 grid-stride) --------------------
__global__ void gcn_zero_f4(float4* __restrict__ p, long n4) {
    long i      = (long)blockIdx.x * blockDim.x + threadIdx.x;
    long stride = (long)gridDim.x * blockDim.x;
    float4 z = make_float4(0.f, 0.f, 0.f, 0.f);
    for (; i < n4; i += stride) p[i] = z;
}

// --- pack W (f32 row-major) -> bf16 fragment-major --------------------------
// Destination index i decodes as {tn, kt, lane, e}; element value is
// W[K][tn*16 + idx] with K = kt*32 + (e<8 ? e+8*half : e+8+8*half),
// half = lane>>4, idx = lane&15. A wave's B-fragment for (tn,kt) is then
// 32 contiguous bytes per lane. tn-major => each N-half is contiguous 64 KB.
__global__ void gcn_pack_w(const float* __restrict__ w,
                           unsigned short* __restrict__ wpack) {
    int i = blockIdx.x * blockDim.x + threadIdx.x;
    if (i >= W_PACK_ELEMS) return;
    int e    = i & 15;
    int lane = (i >> 4) & 31;
    int kt   = (i >> 9) & 7;
    int tn   = i >> 12;
    int half = lane >> 4;
    int idx  = lane & 15;
    int K = kt * 32 + ((e < 8) ? (e + 8 * half) : (e + 8 + 8 * half));
    wpack[i] = f32_to_bf16_rne(w[K * D_FEAT + tn * 16 + idx]);
}

// --- SpMM scatter: one wave32 per edge, 8 floats per lane -------------------
__global__ void gcn_scatter(const float*      __restrict__ x,
                            const long long*  __restrict__ src,
                            const long long*  __restrict__ dst,
                            const float*      __restrict__ val,
                            float*            __restrict__ agg,
                            int nEdges) {
    int wave = (int)((blockIdx.x * (long)blockDim.x + threadIdx.x) >> 5);
    int lane = threadIdx.x & 31;
    if (wave >= nEdges) return;

    long long s = src[wave];
    long long d = dst[wave];
    float     v = val[wave];

    const float4* xr = (const float4*)(x + s * D_FEAT + lane * 8);
    float4 a0 = xr[0];
    float4 a1 = xr[1];

    float* ar = agg + d * D_FEAT + lane * 8;
    // 8 hardware f32 atomics per lane; accumulator (100MB) is L2-resident.
    unsafeAtomicAdd(ar + 0, v * a0.x);
    unsafeAtomicAdd(ar + 1, v * a0.y);
    unsafeAtomicAdd(ar + 2, v * a0.z);
    unsafeAtomicAdd(ar + 3, v * a0.w);
    unsafeAtomicAdd(ar + 4, v * a1.x);
    unsafeAtomicAdd(ar + 5, v * a1.y);
    unsafeAtomicAdd(ar + 6, v * a1.z);
    unsafeAtomicAdd(ar + 7, v * a1.w);
}

// --- WMMA GEMM: one wave = one 16-row strip x 8 column tiles ----------------
// blockIdx.x bit 0 selects the N-half; blockIdx.x>>1 selects the tm group.
__global__ void gcn_gemm_wmma(const float*          __restrict__ agg,
                              const unsigned short* __restrict__ wpack,
                              const float*          __restrict__ bias,
                              float*                __restrict__ out,
                              int nTmTiles) {
    __shared__ unsigned short smemW[W_HALF_ELEMS];   // 64 KB

    const int nh = blockIdx.x & 1;                   // N-half: tn in [nh*8, nh*8+8)

    // Cooperative stage of this half of packed W: 4096 x 16B.
    {
        const uint4* s = (const uint4*)(wpack + nh * W_HALF_ELEMS);
        uint4*       d = (uint4*)smemW;
        for (int i = threadIdx.x; i < W_HALF_ELEMS / 8; i += blockDim.x)
            d[i] = s[i];
    }
    __syncthreads();

    const int lane = threadIdx.x & 31;
    const int half = lane >> 4;      // 0 or 1
    const int idx  = lane & 15;
    const int tm   = (blockIdx.x >> 1) * 8 + (threadIdx.x >> 5);
    if (tm >= nTmTiles) return;      // wave-uniform; HW barrier already passed

    const float* arow = agg + (long)(tm * 16 + idx) * D_FEAT;

    v8f acc[TN_PER_WAVE];
    #pragma unroll
    for (int t = 0; t < TN_PER_WAVE; ++t) acc[t] = (v8f){};

    union { v16bf v; __bf16 b[16]; } afrag;

    for (int kt = 0; kt < N_KSTEPS; ++kt) {
        const int ka = kt * 32 + 8 * half;
        // A: 16 f32 as 4x b128 loads, native RNE casts to bf16.
        float4 f0 = *(const float4*)(arow + ka + 0);
        float4 f1 = *(const float4*)(arow + ka + 4);
        float4 f2 = *(const float4*)(arow + ka + 16);
        float4 f3 = *(const float4*)(arow + ka + 20);
        afrag.b[0]  = (__bf16)f0.x;  afrag.b[1]  = (__bf16)f0.y;
        afrag.b[2]  = (__bf16)f0.z;  afrag.b[3]  = (__bf16)f0.w;
        afrag.b[4]  = (__bf16)f1.x;  afrag.b[5]  = (__bf16)f1.y;
        afrag.b[6]  = (__bf16)f1.z;  afrag.b[7]  = (__bf16)f1.w;
        afrag.b[8]  = (__bf16)f2.x;  afrag.b[9]  = (__bf16)f2.y;
        afrag.b[10] = (__bf16)f2.z;  afrag.b[11] = (__bf16)f2.w;
        afrag.b[12] = (__bf16)f3.x;  afrag.b[13] = (__bf16)f3.y;
        afrag.b[14] = (__bf16)f3.z;  afrag.b[15] = (__bf16)f3.w;

        // 8 column tiles: B fragment = 32 contiguous bytes/lane from LDS.
        #pragma unroll
        for (int t = 0; t < TN_PER_WAVE; ++t) {
            v16bf bfrag = *(const v16bf*)(
                smemW + (((t * N_KSTEPS + kt) * 32 + lane) << 4));
            acc[t] = __builtin_amdgcn_wmma_f32_16x16x32_bf16(
                         /*neg_a=*/false, afrag.v,
                         /*neg_b=*/false, bfrag,
                         /*c_mod=*/(short)0, acc[t],
                         /*reuse_a=*/false, /*reuse_b=*/false);
        }
    }

    // Epilogue: bias + ReLU. Lane -> col (idx), VGPR r -> row r + 8*half.
    #pragma unroll
    for (int t = 0; t < TN_PER_WAVE; ++t) {
        const int tn = nh * TN_PER_WAVE + t;
        float bv = bias[tn * 16 + idx];
        #pragma unroll
        for (int r = 0; r < 8; ++r) {
            long row = (long)tm * 16 + r + 8 * half;
            float o  = acc[t][r] + bv;
            out[row * D_FEAT + tn * 16 + idx] = o > 0.f ? o : 0.f;
        }
    }
}

// ---------------------------------------------------------------------------
extern "C" void kernel_launch(void* const* d_in, const int* in_sizes, int n_in,
                              void* d_out, int out_size, void* d_ws, size_t ws_size,
                              hipStream_t stream) {
    const float*     x    = (const float*)d_in[0];       // [N, 256] f32
    const long long* esrc = (const long long*)d_in[1];   // [E] i64
    const long long* edst = (const long long*)d_in[2];   // [E] i64
    const float*     eval = (const float*)d_in[3];       // [E] f32
    const float*     W    = (const float*)d_in[4];       // [256,256] f32
    const float*     bias = (const float*)d_in[5];       // [256] f32

    const int nNodes = in_sizes[0] / D_FEAT;
    const int nEdges = in_sizes[1];

    // Workspace layout: [agg f32 N*256][W packed bf16 65536]
    float* agg = (float*)d_ws;
    size_t aggBytes = (size_t)nNodes * D_FEAT * sizeof(float);
    unsigned short* wpack = (unsigned short*)((char*)d_ws + aggBytes);

    // 1) zero the accumulator (deterministic per call)
    long n4 = (long)nNodes * (D_FEAT / 4);
    gcn_zero_f4<<<4096, 256, 0, stream>>>((float4*)agg, n4);

    // 2) W -> bf16, fragment-major
    gcn_pack_w<<<W_PACK_ELEMS / 256, 256, 0, stream>>>(W, wpack);

    // 3) edge scatter: one wave32 per edge
    long scatterThreads = (long)nEdges * 32;
    gcn_scatter<<<(unsigned)((scatterThreads + 255) / 256), 256, 0, stream>>>(
        x, esrc, edst, eval, agg, nEdges);

    // 4) WMMA GEMM + bias + ReLU: block = (tm group of 8) x (N-half)
    int nTmTiles = (nNodes + 15) / 16;                   // 6250
    int nBlocks  = 2 * ((nTmTiles + 7) / 8);             // 1564
    gcn_gemm_wmma<<<nBlocks, 256, 0, stream>>>(
        agg, wpack, bias, (float*)d_out, nTmTiles);
}
